// AttentiveAutoEncoder_55645596287496
// MI455X (gfx1250) — matrix operand order repaired and checked by reading, not compiled, not run on hardware
//
#include <hip/hip_runtime.h>
#include <hip/hip_bf16.h>
#include <math.h>

// ---------------------------------------------------------------------------
// AttentiveAutoEncoder forward for MI455X (gfx1250, wave32, WMMA).
// Dominant cost: 8 grouped GEMMs [2048x256]x[256x256] per feature (64 groups)
// ~137 GFLOP -> v_wmma_f32_16x16x32_bf16 with bf16 weights/activations.
// Staging now uses GLOBAL_LOAD_ASYNC_TO_LDS_B128 (ASYNCcnt) instead of
// global->VGPR->ds_store, per CDNA5 ISA ch.10.7/15.18.3.
// ---------------------------------------------------------------------------

typedef __bf16 bf16_t;
typedef __attribute__((ext_vector_type(16))) __bf16 v16bf;
typedef __attribute__((ext_vector_type(8)))  __bf16 v8bf;
typedef __attribute__((ext_vector_type(8)))  float  v8f;

#define B_DIM 2048
#define F_DIM 64
#define E_DIM 8
#define H_DIM 256
#define NH_DIM 4
#define HD_DIM 64
#define OUT_DIM 64
#define NL_DIM 4

#define SW_STRIDE 272   // padded LDS row stride (bf16 elems) to dodge bank conflicts
#define SA_STRIDE 272
#define MTILE 64
#define OM_TILE 128

// Async 16B copy global -> LDS (per enabled lane), tracked with ASYNCcnt.
// GV mode: %0 = LDS byte offset (VGPR), %1 = 64-bit global address (VGPR pair).
__device__ __forceinline__ void async_copy_b128(unsigned ldsOff, const void* gsrc) {
  asm volatile("global_load_async_to_lds_b128 %0, %1, off"
               :: "v"(ldsOff), "v"(gsrc) : "memory");
}
__device__ __forceinline__ void wait_async0() {
  asm volatile("s_wait_asynccnt 0x0" ::: "memory");
}
__device__ __forceinline__ unsigned lds_off(const void* p) {
  // generic LDS pointers carry the wave-relative LDS byte offset in addr[31:0]
  return (unsigned)(size_t)p;
}

// ---------------------------------------------------------------------------
// 1) Weight convert fp32 -> bf16 with transpose to [out][in] so WMMA B-frags
//    are contiguous 16B LDS reads. Dest-coalesced grid-stride loop.
// ---------------------------------------------------------------------------
__global__ __launch_bounds__(256)
void convert_weights(const float* __restrict__ Wenc, const float* __restrict__ Wdec,
                     const float* __restrict__ Wout,
                     bf16_t* __restrict__ wtEnc, bf16_t* __restrict__ wtDec,
                     bf16_t* __restrict__ wtOut) {
  const long ENC_N = (long)F_DIM * NL_DIM * H_DIM * H_DIM;  // 16,777,216
  const long total = 2L * ENC_N + (long)H_DIM * OUT_DIM;
  long stride = (long)gridDim.x * blockDim.x;
  for (long idx = (long)blockIdx.x * blockDim.x + threadIdx.x; idx < total; idx += stride) {
    if (idx < ENC_N) {
      long d = idx;                         // [f][j][o][h] dest
      int h = (int)(d & 255), o = (int)((d >> 8) & 255); long fj = d >> 16;
      wtEnc[d] = (bf16_t)Wenc[(fj << 16) + (long)h * H_DIM + o];   // src [f][j][h][o]
    } else if (idx < 2L * ENC_N) {
      long d = idx - ENC_N;
      int h = (int)(d & 255), o = (int)((d >> 8) & 255); long fj = d >> 16;
      wtDec[d] = (bf16_t)Wdec[(fj << 16) + (long)h * H_DIM + o];
    } else {
      long d = idx - 2L * ENC_N;            // dest [n][h], n in [0,64)
      int h = (int)(d & 255), n = (int)(d >> 8);
      wtOut[d] = (bf16_t)Wout[(long)h * OUT_DIM + n];              // src [h][n]
    }
  }
}

// ---------------------------------------------------------------------------
// 2) Attention constant prep (rank-1 collapse of embed/qkv/in_proj/out_proj).
//    q2[b,f,:] = x[b,f]*uq + cq  etc.  One block, 256 threads (t = channel).
// ---------------------------------------------------------------------------
__global__ __launch_bounds__(256)
void attn_prep(const float* __restrict__ W_emb,
               const float* __restrict__ Wq, const float* __restrict__ bq,
               const float* __restrict__ Wk, const float* __restrict__ bk,
               const float* __restrict__ Wv, const float* __restrict__ bv,
               const float* __restrict__ Win, const float* __restrict__ bin_,
               const float* __restrict__ Wo, const float* __restrict__ bo,
               float* __restrict__ scal /*[NH][4]*/, float* __restrict__ wvh /*[NH][H]*/,
               float* __restrict__ abase /*[H]*/) {
  __shared__ float vqL[H_DIM], vkL[H_DIM], vvL[H_DIM];
  __shared__ float uqL[H_DIM], cqL[H_DIM], ukL[H_DIM], ckL[H_DIM], uvL[H_DIM], cvL[H_DIM];
  int t = threadIdx.x;
  float vq = 0.f, vk = 0.f, vv = 0.f;
  #pragma unroll
  for (int e = 0; e < E_DIM; ++e) {
    float w = W_emb[e];
    vq += w * Wq[e * H_DIM + t];
    vk += w * Wk[e * H_DIM + t];
    vv += w * Wv[e * H_DIM + t];
  }
  vqL[t] = vq; vkL[t] = vk; vvL[t] = vv;
  __syncthreads();
  float uq = 0.f, cq = 0.f, uk = 0.f, ck = 0.f, uv = 0.f, cv = 0.f;
  const float* wq2 = Win + (size_t)t * H_DIM;                 // Win rows [0,256)
  const float* wk2 = Win + (size_t)(H_DIM + t) * H_DIM;       // rows [256,512)
  const float* wv2 = Win + (size_t)(2 * H_DIM + t) * H_DIM;   // rows [512,768)
  for (int h = 0; h < H_DIM; ++h) {
    uq += vqL[h] * wq2[h];  cq += bq[h] * wq2[h];
    uk += vkL[h] * wk2[h];  ck += bk[h] * wk2[h];
    uv += vvL[h] * wv2[h];  cv += bv[h] * wv2[h];
  }
  cq += bin_[t]; ck += bin_[H_DIM + t]; cv += bin_[2 * H_DIM + t];
  uqL[t] = uq; cqL[t] = cq; ukL[t] = uk; ckL[t] = ck; uvL[t] = uv; cvL[t] = cv;
  __syncthreads();
  if (t < NH_DIM) {  // per-head score scalars
    float A = 0.f, Bq = 0.f, Bk = 0.f, C = 0.f;
    for (int d = 0; d < HD_DIM; ++d) {
      int i = t * HD_DIM + d;
      A  += uqL[i] * ukL[i];
      Bq += uqL[i] * ckL[i];
      Bk += cqL[i] * ukL[i];
      C  += cqL[i] * ckL[i];
    }
    scal[t * 4 + 0] = A; scal[t * 4 + 1] = Bq; scal[t * 4 + 2] = Bk; scal[t * 4 + 3] = C;
  }
  const float* wo = Wo + (size_t)t * H_DIM;  // torch layout: a = ctx @ Wo.T
  float ab = bo[t];
  #pragma unroll
  for (int hh = 0; hh < NH_DIM; ++hh) {
    float s = 0.f;
    for (int d = 0; d < HD_DIM; ++d) s += wo[hh * HD_DIM + d] * uvL[hh * HD_DIM + d];
    wvh[hh * H_DIM + t] = s;
  }
  for (int k = 0; k < H_DIM; ++k) ab += wo[k] * cvL[k];
  abase[t] = ab;
}

// ---------------------------------------------------------------------------
// 3) Attention: closed-form scores + softmax + attn write + 'a' activation in
//    feature-major bf16 layout act0[f][b][h]. One block per batch row.
// ---------------------------------------------------------------------------
__global__ __launch_bounds__(256)
void attn_kernel(const float* __restrict__ x, const float* __restrict__ scal,
                 const float* __restrict__ wvh, const float* __restrict__ abase,
                 float* __restrict__ attnOut, bf16_t* __restrict__ act0) {
  __shared__ float xs[F_DIM];
  __shared__ float sArr[NH_DIM][F_DIM];
  int b = blockIdx.x;
  int t = threadIdx.x;
  if (t < F_DIM) xs[t] = x[(size_t)b * F_DIM + t];
  __syncthreads();
  int head = t >> 6;
  int i = t & 63;
  float A  = scal[head * 4 + 0] * 0.125f;   // fold 1/sqrt(HD)=1/8
  float Bq = scal[head * 4 + 1] * 0.125f;
  float Bk = scal[head * 4 + 2] * 0.125f;
  float C  = scal[head * 4 + 3] * 0.125f;
  float xi = xs[i];
  float p[F_DIM];
  float mx = -3.4e38f;
  #pragma unroll
  for (int j = 0; j < F_DIM; ++j) {
    float v = xi * xs[j] * A + xi * Bq + xs[j] * Bk + C;
    p[j] = v;
    mx = fmaxf(mx, v);
  }
  float sum = 0.f;
  #pragma unroll
  for (int j = 0; j < F_DIM; ++j) { p[j] = __expf(p[j] - mx); sum += p[j]; }
  float inv = 1.f / sum;
  float s = 0.f;
  float* dst = attnOut + ((size_t)b * NH_DIM + head) * F_DIM * F_DIM + (size_t)i * F_DIM;
  #pragma unroll
  for (int j = 0; j < F_DIM; ++j) {
    float a = p[j] * inv;
    dst[j] = a;
    s += a * xs[j];                 // ctx collapses to s*uv + cv (rows sum to 1)
  }
  sArr[head][i] = s;
  __syncthreads();
  // phase 2: t == output channel h'; a[b,f,h'] = abase + sum_head s*wvh
  float w0 = wvh[0 * H_DIM + t], w1 = wvh[1 * H_DIM + t];
  float w2 = wvh[2 * H_DIM + t], w3 = wvh[3 * H_DIM + t];
  float ab = abase[t];
  for (int f = 0; f < F_DIM; ++f) {
    float a = ab + sArr[0][f] * w0 + sArr[1][f] * w1 + sArr[2][f] * w2 + sArr[3][f] * w3;
    act0[((size_t)f * B_DIM + b) * H_DIM + t] = (bf16_t)a;   // coalesced over t
  }
}

// ---------------------------------------------------------------------------
// WMMA fragment loaders (bf16, 16x16x32) in a contiguous k order shared by A and B (not the ISA table order).
// A (16x32, M=lane&15): elems 0..15 -> K = kk+16*half .. +15 (contiguous).
// B (32x16, N=lane&15): elems 0..15 -> K = kk+16*half .. +15 (contiguous).
// ---------------------------------------------------------------------------
__device__ __forceinline__ v16bf frag_a(const bf16_t* aRow, int kk, int half) {
  v8bf lo = *(const v8bf*)(aRow + kk + 16 * half);
  v8bf hi = *(const v8bf*)(aRow + kk + 16 * half + 8);
  v16bf r;
  #pragma unroll
  for (int q = 0; q < 8; ++q) { r[q] = lo[q]; r[q + 8] = hi[q]; }
  return r;
}
__device__ __forceinline__ v16bf frag_b(const bf16_t* bRow) {
  v8bf lo = *(const v8bf*)(bRow);
  v8bf hi = *(const v8bf*)(bRow + 8);
  v16bf r;
  #pragma unroll
  for (int q = 0; q < 8; ++q) { r[q] = lo[q]; r[q + 8] = hi[q]; }
  return r;
}

// ---------------------------------------------------------------------------
// 4) Grouped MLP layer: for feature f, z' = relu(z @ W[f,j] + b[f,j]).
//    Block = 256 threads (8 waves), tile 64(M) x 256(N), K=256.
//    Wave grid 4x2: each wave -> 16x128 via 8 WMMA accumulators.
//    LDS (dynamic, ~170KB of the WGP's 320KB): full Wt[f,j] + A tile,
//    staged via async global->LDS DMA (no VGPR round-trip).
// ---------------------------------------------------------------------------
__global__ __launch_bounds__(256)
void mlp_layer(const bf16_t* __restrict__ wt, const float* __restrict__ bias,
               const bf16_t* __restrict__ actIn, bf16_t* __restrict__ actOut, int j) {
  extern __shared__ char smem[];
  bf16_t* sW = (bf16_t*)smem;                 // [256][SW_STRIDE]  Wt[n][k]
  bf16_t* sA = sW + 256 * SW_STRIDE;          // [64][SA_STRIDE]   A[m][k]
  int f = blockIdx.y;
  int mtile = blockIdx.x;
  int t = threadIdx.x;
  const bf16_t* wsrc = wt + (((size_t)f * NL_DIM + j) << 16);
  const float*  bsrc = bias + ((size_t)f * NL_DIM + j) * H_DIM;
  const bf16_t* asrc = actIn + ((size_t)f * B_DIM + (size_t)mtile * MTILE) * H_DIM;
  // stage weights: 256x256 bf16 = 8192 x 16B async chunks
  #pragma unroll
  for (int c0 = 0; c0 < 32; ++c0) {
    int c = t + c0 * 256;
    int row = c >> 5, off = (c & 31) * 8;
    async_copy_b128(lds_off(sW + row * SW_STRIDE + off), wsrc + (size_t)c * 8);
  }
  // stage activations: 64x256 bf16 = 2048 x 16B async chunks
  #pragma unroll
  for (int c0 = 0; c0 < 8; ++c0) {
    int c = t + c0 * 256;
    int row = c >> 5, off = (c & 31) * 8;
    async_copy_b128(lds_off(sA + row * SA_STRIDE + off), asrc + (size_t)c * 8);
  }
  wait_async0();
  __syncthreads();

  int lane = t & 31, wave = t >> 5;
  int half = lane >> 4, l16 = lane & 15;
  int mWave = (wave & 3) * 16;      // 4 waves across M
  int nWave = (wave >> 2) * 128;    // 2 waves across N
  v8f acc[8] = {};
  const bf16_t* aRow = sA + (mWave + l16) * SA_STRIDE;
  #pragma unroll
  for (int kk = 0; kk < H_DIM; kk += 32) {
    v16bf af = frag_a(aRow, kk, half);
    #pragma unroll
    for (int tt = 0; tt < 8; ++tt) {
      const bf16_t* bRow = sW + (nWave + tt * 16 + l16) * SW_STRIDE + kk + 16 * half;
      v16bf bfr = frag_b(bRow);
      acc[tt] = __builtin_amdgcn_wmma_f32_16x16x32_bf16(
          false, af, false, bfr, (short)0, acc[tt], false, false);
    }
  }
  // epilogue: bias + relu + bf16 store; C/D layout: m = r + 8*half, n = lane&15
  size_t outBase = ((size_t)f * B_DIM + (size_t)mtile * MTILE) * H_DIM;
  #pragma unroll
  for (int tt = 0; tt < 8; ++tt) {
    int n = nWave + tt * 16 + l16;
    float bn = bsrc[n];
    #pragma unroll
    for (int r = 0; r < 8; ++r) {
      int m = mWave + r + 8 * half;
      float v = acc[tt][r] + bn;
      v = fmaxf(v, 0.f);
      actOut[outBase + (size_t)m * H_DIM + n] = (bf16_t)v;
    }
  }
}

// ---------------------------------------------------------------------------
// 5) Output projection: sigmoid(z @ Wout + bout) -> d_out fp32 [B][F][OUT].
//    Block tile 128(M) x 64(N), K=256; 8 waves stacked along M, 4 acc each.
// ---------------------------------------------------------------------------
__global__ __launch_bounds__(256)
void out_proj(const bf16_t* __restrict__ wtOut, const float* __restrict__ bout,
              const bf16_t* __restrict__ actIn, float* __restrict__ out) {
  extern __shared__ char smem[];
  bf16_t* sW = (bf16_t*)smem;                 // [64][SW_STRIDE]  WoutT[n][k]
  bf16_t* sA = sW + 64 * SW_STRIDE;           // [128][SA_STRIDE]
  int f = blockIdx.y, mtile = blockIdx.x, t = threadIdx.x;
  const bf16_t* asrc = actIn + ((size_t)f * B_DIM + (size_t)mtile * OM_TILE) * H_DIM;
  #pragma unroll
  for (int c0 = 0; c0 < 8; ++c0) {            // 64x256 bf16 weights
    int c = t + c0 * 256;
    int row = c >> 5, off = (c & 31) * 8;
    async_copy_b128(lds_off(sW + row * SW_STRIDE + off), wtOut + (size_t)c * 8);
  }
  #pragma unroll
  for (int c0 = 0; c0 < 16; ++c0) {           // 128x256 bf16 activations
    int c = t + c0 * 256;
    int row = c >> 5, off = (c & 31) * 8;
    async_copy_b128(lds_off(sA + row * SA_STRIDE + off), asrc + (size_t)c * 8);
  }
  wait_async0();
  __syncthreads();

  int lane = t & 31, wave = t >> 5;
  int half = lane >> 4, l16 = lane & 15;
  int mWave = wave * 16;
  v8f acc[4] = {};
  const bf16_t* aRow = sA + (mWave + l16) * SA_STRIDE;
  #pragma unroll
  for (int kk = 0; kk < H_DIM; kk += 32) {
    v16bf af = frag_a(aRow, kk, half);
    #pragma unroll
    for (int tt = 0; tt < 4; ++tt) {
      const bf16_t* bRow = sW + (tt * 16 + l16) * SW_STRIDE + kk + 16 * half;
      v16bf bfr = frag_b(bRow);
      acc[tt] = __builtin_amdgcn_wmma_f32_16x16x32_bf16(
          false, af, false, bfr, (short)0, acc[tt], false, false);
    }
  }
  int bBase = mtile * OM_TILE;
  #pragma unroll
  for (int tt = 0; tt < 4; ++tt) {
    int n = tt * 16 + l16;
    float bn = bout[n];
    #pragma unroll
    for (int r = 0; r < 8; ++r) {
      int m = mWave + r + 8 * half;
      float v = acc[tt][r] + bn;
      v = 1.f / (1.f + __expf(-v));
      out[(size_t)(bBase + m) * (F_DIM * OUT_DIM) + (size_t)f * OUT_DIM + n] = v;
    }
  }
}

// ---------------------------------------------------------------------------
// Launch
// ---------------------------------------------------------------------------
extern "C" void kernel_launch(void* const* d_in, const int* in_sizes, int n_in,
                              void* d_out, int out_size, void* d_ws, size_t ws_size,
                              hipStream_t stream) {
  const float* x    = (const float*)d_in[0];
  const float* W_emb= (const float*)d_in[1];
  const float* Wq   = (const float*)d_in[2];
  const float* bq   = (const float*)d_in[3];
  const float* Wk   = (const float*)d_in[4];
  const float* bk   = (const float*)d_in[5];
  const float* Wv   = (const float*)d_in[6];
  const float* bv   = (const float*)d_in[7];
  const float* Win  = (const float*)d_in[8];
  const float* bin_ = (const float*)d_in[9];
  const float* Wo   = (const float*)d_in[10];
  const float* bo   = (const float*)d_in[11];
  const float* Wenc = (const float*)d_in[12];
  const float* benc = (const float*)d_in[13];
  const float* Wdec = (const float*)d_in[14];
  const float* bdec = (const float*)d_in[15];
  const float* Wout = (const float*)d_in[16];
  const float* bout = (const float*)d_in[17];

  float* out = (float*)d_out;                                       // [B][F][OUT]
  float* attnOut = out + (size_t)B_DIM * F_DIM * OUT_DIM;           // [B][NH][F][F]

  char* ws = (char*)d_ws;
  bf16_t* wtEnc = (bf16_t*)(ws);                      // 33,554,432 B
  bf16_t* wtDec = (bf16_t*)(ws + 33554432ull);        // 33,554,432 B
  bf16_t* wtOutW= (bf16_t*)(ws + 67108864ull);        //     32,768 B
  bf16_t* act0  = (bf16_t*)(ws + 67141632ull);        // 67,108,864 B
  bf16_t* act1  = (bf16_t*)(ws + 134250496ull);       // 67,108,864 B
  float*  scal  = (float*)(ws + 201359360ull);        // 64 B
  float*  wvh   = (float*)(ws + 201359360ull + 256);  // 4 KB
  float*  abase = (float*)(ws + 201359360ull + 256 + 4096);  // 1 KB

  convert_weights<<<4096, 256, 0, stream>>>(Wenc, Wdec, Wout, wtEnc, wtDec, wtOutW);
  attn_prep<<<1, 256, 0, stream>>>(W_emb, Wq, bq, Wk, bk, Wv, bv, Win, bin_, Wo, bo,
                                   scal, wvh, abase);
  attn_kernel<<<B_DIM, 256, 0, stream>>>(x, scal, wvh, abase, attnOut, act0);

  size_t layerLds = (size_t)(256 * SW_STRIDE + 64 * SA_STRIDE) * sizeof(bf16_t);
  dim3 grid(B_DIM / MTILE, F_DIM);
  bf16_t* cur = act0;
  bf16_t* nxt = act1;
  for (int j = 0; j < NL_DIM; ++j) {
    mlp_layer<<<grid, 256, layerLds, stream>>>(wtEnc, benc, cur, nxt, j);
    bf16_t* tmp = cur; cur = nxt; nxt = tmp;
  }
  for (int j = 0; j < NL_DIM; ++j) {
    mlp_layer<<<grid, 256, layerLds, stream>>>(wtDec, bdec, cur, nxt, j);
    bf16_t* tmp = cur; cur = nxt; nxt = tmp;
  }
  size_t outLds = (size_t)(64 * SW_STRIDE + 128 * SA_STRIDE) * sizeof(bf16_t);
  dim3 ogrid(B_DIM / OM_TILE, F_DIM);
  out_proj<<<ogrid, 256, outLds, stream>>>(wtOutW, bout, cur, out);
}
